// SwinTransformerBlock_59751585022227
// MI455X (gfx1250) — matrix operand order repaired and checked by reading, not compiled, not run on hardware
//
#include <hip/hip_runtime.h>
#include <hip/hip_bf16.h>
#include <math.h>

typedef __attribute__((ext_vector_type(16))) _Float16 v16h;
typedef __attribute__((ext_vector_type(8)))  _Float16 v8h;
typedef __attribute__((ext_vector_type(8)))  float    v8f;

#define DIM   512
#define HEADS 16
#define DH    32
#define WS    7
#define SHIFTV 3
#define HH    56
#define NWIN  1024          // B * 64 windows
#define NTOK  64            // padded tokens per window (49 real)
#define MQ    (NWIN * NTOK) // 65536 padded token rows
#define MX    (16 * 56 * 56) // 50176 real token rows
#define HID   2048

// ---------------------------------------------------------------------------
// WMMA fragment loaders (CDNA5 16-bit layouts, wave32)
// A 16x32: lane L: row M = L&15 ; halves 0-7 -> K = 8*(L>>4)+j ; 8-15 -> K = 16+8*(L>>4)+(j-8)
// B 32x16: lane L: col N = L&15 ; halves 0-7 -> K = 8*(L>>4)+j ; 8-15 -> K = 16+8*(L>>4)+(j-8)
// ---------------------------------------------------------------------------
__device__ inline v16h frag_a(const _Float16* rowbase, int hi) {
    const int ko = hi * 8;
    v8h lo = *(const v8h*)(rowbase + ko);
    v8h hv = *(const v8h*)(rowbase + 16 + ko);
    v16h r;
#pragma unroll
    for (int i = 0; i < 8; ++i) { r[i] = lo[i]; r[i + 8] = hv[i]; }
    return r;
}

__device__ inline v16h frag_b(const _Float16* rowbase, int hi) {
    const int ko = hi * 8;
    v8h lo = *(const v8h*)(rowbase + ko);
    v8h hv = *(const v8h*)(rowbase + 16 + ko);
    v16h r;
#pragma unroll
    for (int i = 0; i < 8; ++i) { r[i] = lo[i]; r[i + 8] = hv[i]; }
    return r;
}

__device__ inline v8f wmma_f16(v16h a, v16h b, v8f c) {
    return __builtin_amdgcn_wmma_f32_16x16x32_f16(false, a, false, b, (short)0, c, false, false);
}

// ---------------------------------------------------------------------------
// Shared tiled GEMM core: C[M,N] = A[M,K] * W[N,K]^T  (both f16, f32 accum)
// block = 256 threads (8 waves), block tile 128x256, wave grid 2(M) x 4(N),
// wave tile 64x64 = 4x4 accumulator tiles, K-step 32.
// Per K-step per wave: 8 fragment loads (16 ds_load_b128) -> 16 WMMAs.
// M multiple of 128, N multiple of 256 (true for all four GEMMs here).
// ---------------------------------------------------------------------------
template <class Epi>
__device__ inline void gemm_tile(const _Float16* __restrict__ A,
                                 const _Float16* __restrict__ W,
                                 int K, _Float16* lA, _Float16* lB, Epi&& epi) {
    const int tid  = threadIdx.x;
    const int lane = tid & 31, wave = tid >> 5;
    const int wm = wave & 1, wn = wave >> 1;          // 2 x 4 wave grid
    const int m0 = blockIdx.y * 128, n0 = blockIdx.x * 256;
    const int lm = lane & 15, hi = lane >> 4;

    v8f acc[4][4] = {};

    for (int kb = 0; kb < K; kb += 32) {
        // cooperative copies into LDS (stride 40 halves = 80B, 16B aligned)
        // A: 128x32 halves = 512 v8h ; B: 256x32 halves = 1024 v8h
#pragma unroll
        for (int j = 0; j < 2; ++j) {
            int idx = tid + j * 256;
            int row = idx >> 2, qd = idx & 3;
            const _Float16* ga = A + (size_t)(m0 + row) * K + kb + qd * 8;
            *(v8h*)(lA + row * 40 + qd * 8) = *(const v8h*)ga;
            if (kb + 32 < K) __builtin_prefetch(ga + 32, 0, 0);
        }
#pragma unroll
        for (int j = 0; j < 4; ++j) {
            int idx = tid + j * 256;
            int row = idx >> 2, qd = idx & 3;
            const _Float16* gb = W + (size_t)(n0 + row) * K + kb + qd * 8;
            *(v8h*)(lB + row * 40 + qd * 8) = *(const v8h*)gb;
            if (kb + 32 < K) __builtin_prefetch(gb + 32, 0, 0);
        }
        __syncthreads();

        v16h af[4], bf[4];
#pragma unroll
        for (int mt = 0; mt < 4; ++mt)
            af[mt] = frag_a(lA + (wm * 64 + mt * 16 + lm) * 40, hi);
#pragma unroll
        for (int nt = 0; nt < 4; ++nt)
            bf[nt] = frag_b(lB + (wn * 64 + nt * 16 + lm) * 40, hi);
#pragma unroll
        for (int mt = 0; mt < 4; ++mt)
#pragma unroll
            for (int nt = 0; nt < 4; ++nt)
                acc[mt][nt] = wmma_f16(af[mt], bf[nt], acc[mt][nt]);
        __syncthreads();
    }

#pragma unroll
    for (int mt = 0; mt < 4; ++mt)
#pragma unroll
        for (int nt = 0; nt < 4; ++nt)
#pragma unroll
            for (int r = 0; r < 8; ++r) {
                int gm = m0 + wm * 64 + mt * 16 + r + 8 * hi;
                int gn = n0 + wn * 64 + nt * 16 + lm;
                epi(gm, gn, acc[mt][nt][r]);
            }
}

#define GEMM_LDS  __shared__ alignas(16) _Float16 lA[128 * 40], lB[256 * 40]

// ---------------------------------------------------------------------------
// f32 -> f16 conversion
// ---------------------------------------------------------------------------
__global__ __launch_bounds__(256) void k_cvt(const float* __restrict__ s,
                                             _Float16* __restrict__ d, int n) {
    int i = blockIdx.x * 256 + threadIdx.x;
    if (i < n) d[i] = (_Float16)s[i];
}

// ---------------------------------------------------------------------------
// combined bias: bias2[h][n][m] = 2*rp_bias + (-1e9 for padded key columns)
// ---------------------------------------------------------------------------
__global__ __launch_bounds__(256) void k_bias2(const float* __restrict__ table,
                                               const int* __restrict__ rp,
                                               float* __restrict__ bias2) {
    int idx = blockIdx.x * 256 + threadIdx.x;   // 16*64*64 = 65536
    int h = idx >> 12, rem = idx & 4095;
    int n = rem >> 6, m = rem & 63;
    float v;
    if (m >= 49)      v = -1e9f;
    else if (n >= 49) v = 0.f;
    else              v = 2.f * table[rp[n * 49 + m] * HEADS + h];
    bias2[idx] = v;
}

// ---------------------------------------------------------------------------
// LN1 + shift(-3,-3) + window partition -> f16 xw[win][64][512], pads zeroed
// one block per (win,tok), 256 threads, 512 channels
// ---------------------------------------------------------------------------
__global__ __launch_bounds__(256) void k_ln1(const float* __restrict__ x,
                                             const float* __restrict__ w,
                                             const float* __restrict__ b,
                                             _Float16* __restrict__ xw) {
    const int bid = blockIdx.x, tid = threadIdx.x;
    const int tok = bid & 63, win = bid >> 6;
    _Float16* dst = xw + (size_t)bid * DIM;
    if (tok >= 49) { dst[tid] = (_Float16)0.f; dst[tid + 256] = (_Float16)0.f; return; }
    const int bb = win >> 6, wi = win & 63;
    const int wh = wi >> 3, ww = wi & 7, th = tok / 7, tw = tok % 7;
    const int hh = (wh * 7 + th + SHIFTV) % HH, wc = (ww * 7 + tw + SHIFTV) % HH;
    const float* src = x + ((size_t)bb * 3136 + hh * HH + wc) * DIM;
    float v0 = src[tid], v1 = src[tid + 256];
    float s = v0 + v1, ss = v0 * v0 + v1 * v1;
#pragma unroll
    for (int m = 1; m < 32; m <<= 1) { s += __shfl_xor(s, m, 32); ss += __shfl_xor(ss, m, 32); }
    __shared__ float shs[8], shq[8];
    int wv = tid >> 5, ln = tid & 31;
    if (ln == 0) { shs[wv] = s; shq[wv] = ss; }
    __syncthreads();
    if (wv == 0) {
        float a = (ln < 8) ? shs[ln] : 0.f, a2 = (ln < 8) ? shq[ln] : 0.f;
#pragma unroll
        for (int m = 1; m < 8; m <<= 1) { a += __shfl_xor(a, m, 32); a2 += __shfl_xor(a2, m, 32); }
        if (ln == 0) { shs[0] = a; shq[0] = a2; }
    }
    __syncthreads();
    float mean = shs[0] * (1.f / DIM);
    float var  = shq[0] * (1.f / DIM) - mean * mean;
    float inv  = rsqrtf(var + 1e-5f);
    dst[tid]       = (_Float16)((v0 - mean) * inv * w[tid] + b[tid]);
    dst[tid + 256] = (_Float16)((v1 - mean) * inv * w[tid + 256] + b[tid + 256]);
}

// LN2 on residual stream x1 -> h2 (f16), one block per row
__global__ __launch_bounds__(256) void k_ln2(const float* __restrict__ x1,
                                             const float* __restrict__ w,
                                             const float* __restrict__ b,
                                             _Float16* __restrict__ h2) {
    const int tid = threadIdx.x;
    const float* src = x1 + (size_t)blockIdx.x * DIM;
    _Float16* dst = h2 + (size_t)blockIdx.x * DIM;
    float v0 = src[tid], v1 = src[tid + 256];
    float s = v0 + v1, ss = v0 * v0 + v1 * v1;
#pragma unroll
    for (int m = 1; m < 32; m <<= 1) { s += __shfl_xor(s, m, 32); ss += __shfl_xor(ss, m, 32); }
    __shared__ float shs[8], shq[8];
    int wv = tid >> 5, ln = tid & 31;
    if (ln == 0) { shs[wv] = s; shq[wv] = ss; }
    __syncthreads();
    if (wv == 0) {
        float a = (ln < 8) ? shs[ln] : 0.f, a2 = (ln < 8) ? shq[ln] : 0.f;
#pragma unroll
        for (int m = 1; m < 8; m <<= 1) { a += __shfl_xor(a, m, 32); a2 += __shfl_xor(a2, m, 32); }
        if (ln == 0) { shs[0] = a; shq[0] = a2; }
    }
    __syncthreads();
    float mean = shs[0] * (1.f / DIM);
    float var  = shq[0] * (1.f / DIM) - mean * mean;
    float inv  = rsqrtf(var + 1e-5f);
    dst[tid]       = (_Float16)((v0 - mean) * inv * w[tid] + b[tid]);
    dst[tid + 256] = (_Float16)((v1 - mean) * inv * w[tid + 256] + b[tid + 256]);
}

// ---------------------------------------------------------------------------
// GEMM 1: QKV projection. Scatter epilogue -> q(scaled)/k row-major, v transposed
// ---------------------------------------------------------------------------
__global__ __launch_bounds__(256) void k_gemm_qkv(const _Float16* __restrict__ A,
                                                  const _Float16* __restrict__ W,
                                                  const float* __restrict__ bias,
                                                  _Float16* __restrict__ qb,
                                                  _Float16* __restrict__ kb,
                                                  _Float16* __restrict__ vT) {
    GEMM_LDS;
    const float scale = 0.17677669529663687f;  // 32^-0.5
    gemm_tile(A, W, DIM, lA, lB, [&](int gm, int gn, float v) {
        v += bias[gn];
        int h = gn / 96, rem = gn % 96, d = rem / 3, c = rem % 3;
        int win = gm >> 6, tok = gm & 63;
        size_t wh = (size_t)(win * HEADS + h);
        if (c == 0)      qb[(wh * NTOK + tok) * DH + d] = (_Float16)(v * scale);
        else if (c == 1) kb[(wh * NTOK + tok) * DH + d] = (_Float16)v;
        else             vT[(wh * DH + d) * NTOK + tok] = (_Float16)v;
    });
}

// ---------------------------------------------------------------------------
// Attention: one wave per (window, head). 64x64 padded scores via WMMA,
// softmax with 16-lane shuffles, P through LDS, P*V via WMMA.
// ---------------------------------------------------------------------------
__global__ __launch_bounds__(128) void k_attn(const _Float16* __restrict__ qb,
                                              const _Float16* __restrict__ kb,
                                              const _Float16* __restrict__ vT,
                                              const float* __restrict__ bias2,
                                              const float* __restrict__ mask,
                                              _Float16* __restrict__ ob) {
    __shared__ alignas(16) _Float16 lP[4][64 * 72];
    const int tid = threadIdx.x, lane = tid & 31, wv = tid >> 5;
    const int whid = blockIdx.x * 4 + wv;
    const int win = whid >> 4, h = whid & 15;
    const int wi = win & 63;
    const _Float16* qp = qb + (size_t)whid * NTOK * DH;
    const _Float16* kp = kb + (size_t)whid * NTOK * DH;
    const _Float16* vp = vT + (size_t)whid * DH * NTOK;
    _Float16* P = lP[wv];
    const int lm = lane & 15, hi = lane >> 4;
    const float* b2 = bias2 + (size_t)h * 64 * 64;

    v16h bk[4];
#pragma unroll
    for (int tj = 0; tj < 4; ++tj)
        bk[tj] = frag_b(kp + (tj * 16 + lm) * DH, hi);   // B = K^T

#pragma unroll
    for (int ti = 0; ti < 4; ++ti) {
        v16h aq = frag_a(qp + (ti * 16 + lm) * DH, hi);
        v8f sc[4] = {};
#pragma unroll
        for (int tj = 0; tj < 4; ++tj) sc[tj] = wmma_f16(aq, bk[tj], sc[tj]);
        // + 2*rel-pos bias (+pad) + shifted-window mask
#pragma unroll
        for (int tj = 0; tj < 4; ++tj)
#pragma unroll
            for (int r = 0; r < 8; ++r) {
                int n = ti * 16 + r + 8 * hi;
                int m = tj * 16 + lm;
                float add = b2[n * 64 + m];
                if (n < 49 && m < 49) add += mask[((size_t)wi * 49 + n) * 49 + m];
                sc[tj][r] += add;
            }
        // row softmax (row spread across 16 lanes x 4 tiles)
#pragma unroll
        for (int r = 0; r < 8; ++r) {
            float mx = fmaxf(fmaxf(sc[0][r], sc[1][r]), fmaxf(sc[2][r], sc[3][r]));
#pragma unroll
            for (int d = 1; d < 16; d <<= 1) mx = fmaxf(mx, __shfl_xor(mx, d, 32));
            float e0 = __expf(sc[0][r] - mx), e1 = __expf(sc[1][r] - mx);
            float e2 = __expf(sc[2][r] - mx), e3 = __expf(sc[3][r] - mx);
            float s = e0 + e1 + e2 + e3;
#pragma unroll
            for (int d = 1; d < 16; d <<= 1) s += __shfl_xor(s, d, 32);
            float inv = 1.f / (s + 1e-20f);
            int n = ti * 16 + r + 8 * hi;
            P[n * 72 + 0 * 16 + lm] = (_Float16)(e0 * inv);
            P[n * 72 + 1 * 16 + lm] = (_Float16)(e1 * inv);
            P[n * 72 + 2 * 16 + lm] = (_Float16)(e2 * inv);
            P[n * 72 + 3 * 16 + lm] = (_Float16)(e3 * inv);
        }
    }
    __syncthreads();

    v16h bv[2][2];
#pragma unroll
    for (int dj = 0; dj < 2; ++dj)
#pragma unroll
        for (int kk = 0; kk < 2; ++kk)
            bv[dj][kk] = frag_b(vp + (dj * 16 + lm) * NTOK + kk * 32, hi);

#pragma unroll
    for (int ti = 0; ti < 4; ++ti) {
        v16h ap0 = frag_a(P + (ti * 16 + lm) * 72, hi);
        v16h ap1 = frag_a(P + (ti * 16 + lm) * 72 + 32, hi);
#pragma unroll
        for (int dj = 0; dj < 2; ++dj) {
            v8f oc = {};
            oc = wmma_f16(ap0, bv[dj][0], oc);
            oc = wmma_f16(ap1, bv[dj][1], oc);
#pragma unroll
            for (int r = 0; r < 8; ++r) {
                int n = ti * 16 + r + 8 * hi;
                int d = dj * 16 + lm;
                ob[((size_t)win * NTOK + n) * DIM + h * DH + d] = (_Float16)oc[r];
            }
        }
    }
}

// ---------------------------------------------------------------------------
// GEMM 2: output projection + window reverse + roll back + residual -> x1 (f32)
// ---------------------------------------------------------------------------
__global__ __launch_bounds__(256) void k_gemm_proj(const _Float16* __restrict__ A,
                                                   const _Float16* __restrict__ W,
                                                   const float* __restrict__ bias,
                                                   const float* __restrict__ xin,
                                                   float* __restrict__ x1) {
    GEMM_LDS;
    gemm_tile(A, W, DIM, lA, lB, [&](int gm, int gn, float v) {
        int win = gm >> 6, tok = gm & 63;
        if (tok >= 49) return;
        int bb = win >> 6, wi = win & 63;
        int wh = wi >> 3, ww = wi & 7, th = tok / 7, tw = tok % 7;
        int hh = (wh * 7 + th + SHIFTV) % HH, wc = (ww * 7 + tw + SHIFTV) % HH;
        size_t idx = ((size_t)bb * 3136 + hh * HH + wc) * DIM + gn;
        x1[idx] = v + bias[gn] + xin[idx];
    });
}

// GEMM 3: fc1 + exact GELU -> g (f16)
__global__ __launch_bounds__(256) void k_gemm_fc1(const _Float16* __restrict__ A,
                                                  const _Float16* __restrict__ W,
                                                  const float* __restrict__ bias,
                                                  _Float16* __restrict__ g) {
    GEMM_LDS;
    gemm_tile(A, W, DIM, lA, lB, [&](int gm, int gn, float v) {
        float t = v + bias[gn];
        float gl = 0.5f * t * (1.f + erff(t * 0.70710678118654752f));
        g[(size_t)gm * HID + gn] = (_Float16)gl;
    });
}

// GEMM 4: fc2 + bias + residual -> d_out (f32)
__global__ __launch_bounds__(256) void k_gemm_fc2(const _Float16* __restrict__ A,
                                                  const _Float16* __restrict__ W,
                                                  const float* __restrict__ bias,
                                                  const float* __restrict__ x1,
                                                  float* __restrict__ out) {
    GEMM_LDS;
    gemm_tile(A, W, HID, lA, lB, [&](int gm, int gn, float v) {
        size_t idx = (size_t)gm * DIM + gn;
        out[idx] = v + bias[gn] + x1[idx];
    });
}

// ---------------------------------------------------------------------------
extern "C" void kernel_launch(void* const* d_in, const int* in_sizes, int n_in,
                              void* d_out, int out_size, void* d_ws, size_t ws_size,
                              hipStream_t stream) {
    const float* x        = (const float*)d_in[0];
    const float* qkv_w    = (const float*)d_in[1];
    const float* qkv_b    = (const float*)d_in[2];
    const float* out_w    = (const float*)d_in[3];
    const float* out_b    = (const float*)d_in[4];
    const float* rp_table = (const float*)d_in[5];
    const float* fc1_w    = (const float*)d_in[6];
    const float* fc1_b    = (const float*)d_in[7];
    const float* fc2_w    = (const float*)d_in[8];
    const float* fc2_b    = (const float*)d_in[9];
    const float* ln1_w    = (const float*)d_in[10];
    const float* ln1_b    = (const float*)d_in[11];
    const float* ln2_w    = (const float*)d_in[12];
    const float* ln2_b    = (const float*)d_in[13];
    const float* amask    = (const float*)d_in[14];
    const int*   rp_index = (const int*)d_in[15];
    float* out = (float*)d_out;

    char* cur = (char*)d_ws;
    auto alloc = [&](size_t bytes) {
        void* p = (void*)cur;
        cur += (bytes + 255) & ~(size_t)255;
        return p;
    };
    _Float16* wqkv = (_Float16*)alloc((size_t)3 * DIM * DIM * 2);
    _Float16* wout = (_Float16*)alloc((size_t)DIM * DIM * 2);
    _Float16* wfc1 = (_Float16*)alloc((size_t)HID * DIM * 2);
    _Float16* wfc2 = (_Float16*)alloc((size_t)DIM * HID * 2);
    _Float16* xw   = (_Float16*)alloc((size_t)MQ * DIM * 2);
    _Float16* qb   = (_Float16*)alloc((size_t)MQ * DIM * 2);
    _Float16* kb   = (_Float16*)alloc((size_t)MQ * DIM * 2);
    _Float16* vT   = (_Float16*)alloc((size_t)MQ * DIM * 2);
    _Float16* ob   = (_Float16*)alloc((size_t)MQ * DIM * 2);
    float*    x1   = (float*)alloc((size_t)MX * DIM * 4);
    _Float16* h2   = (_Float16*)alloc((size_t)MX * DIM * 2);
    _Float16* g    = (_Float16*)alloc((size_t)MX * HID * 2);
    float*    b2   = (float*)alloc((size_t)HEADS * 64 * 64 * 4);

    // weight conversions f32 -> f16
    int nqkv = 3 * DIM * DIM, nout = DIM * DIM, nfc1 = HID * DIM, nfc2 = DIM * HID;
    k_cvt<<<(nqkv + 255) / 256, 256, 0, stream>>>(qkv_w, wqkv, nqkv);
    k_cvt<<<(nout + 255) / 256, 256, 0, stream>>>(out_w, wout, nout);
    k_cvt<<<(nfc1 + 255) / 256, 256, 0, stream>>>(fc1_w, wfc1, nfc1);
    k_cvt<<<(nfc2 + 255) / 256, 256, 0, stream>>>(fc2_w, wfc2, nfc2);

    // combined rel-pos bias (+pad masking)
    k_bias2<<<HEADS * 64 * 64 / 256, 256, 0, stream>>>(rp_table, rp_index, b2);

    // LN1 + shift + window partition (padded to 64 tokens)
    k_ln1<<<MQ, 256, 0, stream>>>(x, ln1_w, ln1_b, xw);

    // QKV projection: 65536 x 512 x 1536
    k_gemm_qkv<<<dim3(1536 / 256, MQ / 128), 256, 0, stream>>>(xw, wqkv, qkv_b, qb, kb, vT);

    // attention: 16384 window-heads, 4 per block
    k_attn<<<NWIN * HEADS / 4, 128, 0, stream>>>(qb, kb, vT, b2, amask, ob);

    // output projection + reverse + residual: 65536 x 512 x 512
    k_gemm_proj<<<dim3(DIM / 256, MQ / 128), 256, 0, stream>>>(ob, wout, out_b, x, x1);

    // LN2
    k_ln2<<<MX, 256, 0, stream>>>(x1, ln2_w, ln2_b, h2);

    // MLP: 50176 x 512 x 2048, then 50176 x 2048 x 512
    k_gemm_fc1<<<dim3(HID / 256, MX / 128), 256, 0, stream>>>(h2, wfc1, fc1_b, g);
    k_gemm_fc2<<<dim3(DIM / 256, MX / 128), 256, 0, stream>>>(g, wfc2, fc2_b, x1, out);
}
